// TransformerBlock_41601053229694
// MI455X (gfx1250) — matrix operand order repaired and checked by reading, not compiled, not run on hardware
//
#include <hip/hip_runtime.h>
#include <hip/hip_bf16.h>

// ---------------------------------------------------------------------------
// Transformer block for MI455X (gfx1250, wave32, WMMA).
// bf16 WMMA (f32 accumulate) for all GEMMs and both attention matmuls.
// ---------------------------------------------------------------------------

typedef __bf16 bf16_t;
typedef bf16_t v16bf __attribute__((ext_vector_type(16)));
typedef float  v8f   __attribute__((ext_vector_type(8)));

#define DEV static __device__ __forceinline__

// Shapes (fixed by the reference)
constexpr int BS  = 2;
constexpr int SEQ = 2048;
constexpr int DM  = 1024;
constexpr int NH  = 8;
constexpr int HD  = 128;
constexpr int FFN = 4096;
constexpr int MTOT = BS * SEQ;          // 4096 token rows
constexpr int QKVD = 3 * DM;            // 3072

// float -> bf16 (round to nearest even), integer path.
DEV unsigned short f2bf(float x) {
  union { float f; unsigned int u; } c; c.f = x;
  unsigned int r = c.u + 0x7fffu + ((c.u >> 16) & 1u);
  return (unsigned short)(r >> 16);
}
DEV unsigned int pk2(float lo, float hi) {
  return (unsigned int)f2bf(lo) | ((unsigned int)f2bf(hi) << 16);
}

// 16 bf16 per lane (8 VGPRs) = one WMMA A or B operand, loadable as 2x b128.
struct F16x { union { v16bf v; uint4 q[2]; }; };

DEV v8f wmma_bf16(const F16x& a, const F16x& b, v8f c) {
  // (neg_a, A, neg_b, B, c_mod, C, reuse_a, reuse_b)
  return __builtin_amdgcn_wmma_f32_16x16x32_bf16(false, a.v, false, b.v,
                                                 (short)0, c, false, false);
}

// ---------------------------------------------------------------------------
// Generic tiled GEMM: C[M,N] = A[M,K] * B[K,N]  (+ epilogue)
// Block: 256 threads (8 waves). Block tile 128(M) x 256(N), K-step 32.
// Waves arranged 2x4; each wave owns a 64x64 slab = 4x4 WMMA tiles, giving
// 16 ds_load_b128 per 16 WMMAs (1.0 b128/WMMA vs 1.5 for a 32x64 slab) --
// the LDS read path, not the matrix pipes, is the binding resource.
// LDS: A as [M][K] bf16, B transposed as [N][K] bf16 (so B fragments are two
// contiguous b128s), rows padded to 40 halves, double buffered (60 KB).
// EPI: 0 = none, 2 = +bias then exact GELU, 3 = +bias then +residual
// ---------------------------------------------------------------------------
template <int EPI>
__global__ __launch_bounds__(256)
void gemm_wmma_bf16(const float* __restrict__ A, const float* __restrict__ B,
                    const float* __restrict__ bias, const float* __restrict__ resid,
                    float* __restrict__ C, int M, int N, int K)
{
  __shared__ unsigned short aT[2][128][40];   // [buf][m][k]   20480 B
  __shared__ unsigned short bT[2][256][40];   // [buf][n][k]   40960 B

  const int tid  = threadIdx.x;
  const int wave = tid >> 5;
  const int lane = tid & 31;
  const int l15  = lane & 15;
  const bool hi  = lane >= 16;
  const int wm   = wave & 1;           // 64-row slab
  const int wn   = wave >> 1;          // 64-col slab
  const int m0   = blockIdx.y * 128;
  const int n0   = blockIdx.x * 256;

  v8f acc[4][4];
#pragma unroll
  for (int a = 0; a < 4; ++a)
#pragma unroll
    for (int b = 0; b < 4; ++b)
#pragma unroll
      for (int r = 0; r < 8; ++r) acc[a][b][r] = 0.0f;

  float4 ar[4], br[8];

  auto loadTiles = [&](int kt) {
    const float* Ap = A + (size_t)m0 * K + (size_t)kt * 32;
    const float* Bp = B + (size_t)kt * 32 * N + n0;
#pragma unroll
    for (int i = 0; i < 4; ++i) {            // A: 128x32 = 1024 float4
      const int f = tid + 256 * i;
      ar[i] = *(const float4*)(Ap + (size_t)(f >> 3) * K + (f & 7) * 4);
    }
#pragma unroll
    for (int i = 0; i < 8; ++i) {            // B: 32x256 = 2048 float4
      const int f = tid + 256 * i;
      br[i] = *(const float4*)(Bp + (size_t)(f >> 6) * N + (f & 63) * 4);
    }
  };
  auto prefetchTiles = [&](int kt) {         // warm L2 (global_prefetch_b8)
    const float* Ap = A + (size_t)m0 * K + (size_t)kt * 32;
    const float* Bp = B + (size_t)kt * 32 * N + n0;
    __builtin_prefetch(Ap + (size_t)(tid >> 3) * K + (tid & 7) * 4, 0, 3);
    __builtin_prefetch(Bp + (size_t)(tid >> 6) * N + (tid & 63) * 4, 0, 3);
  };
  auto storeTiles = [&](int buf) {
#pragma unroll
    for (int i = 0; i < 4; ++i) {
      const int f = tid + 256 * i;
      const int row = f >> 3, kq = (f & 7) * 4;
      uint2 p; p.x = pk2(ar[i].x, ar[i].y); p.y = pk2(ar[i].z, ar[i].w);
      *(uint2*)&aT[buf][row][kq] = p;        // ds_store_b64
    }
#pragma unroll
    for (int i = 0; i < 8; ++i) {            // transpose B into [n][k]
      const int f = tid + 256 * i;
      const int kr = f >> 6, nq = (f & 63) * 4;
      bT[buf][nq + 0][kr] = f2bf(br[i].x);
      bT[buf][nq + 1][kr] = f2bf(br[i].y);
      bT[buf][nq + 2][kr] = f2bf(br[i].z);
      bT[buf][nq + 3][kr] = f2bf(br[i].w);
    }
  };
  auto compute = [&](int buf) {
    F16x fa[4];
#pragma unroll
    for (int a = 0; a < 4; ++a) {
      // A layout (16-bit 16x32): lanes 0-15 rows M=0..15 K{0..7,16..23},
      // lanes 16-31 same rows K{8..15,24..31}.
      const int row = wm * 64 + a * 16 + l15;
      const int k0  = hi ? 8 : 0;
      fa[a].q[0] = *(const uint4*)&aT[buf][row][k0];
      fa[a].q[1] = *(const uint4*)&aT[buf][row][k0 + 16];
    }
#pragma unroll
    for (int b = 0; b < 4; ++b) {
      // B layout (16-bit 32x16): lanes 0-15 col N, K{0..7,16..23}; lanes 16-31 K{8..15,24..31}.
      F16x fb;
      const int col = wn * 64 + b * 16 + l15;
      const int k0  = hi ? 8 : 0;
      fb.q[0] = *(const uint4*)&bT[buf][col][k0];
      fb.q[1] = *(const uint4*)&bT[buf][col][k0 + 16];
#pragma unroll
      for (int a = 0; a < 4; ++a)
        acc[a][b] = wmma_bf16(fa[a], fb, acc[a][b]);
    }
  };

  const int nk = K >> 5;
  loadTiles(0);
  storeTiles(0);
  __syncthreads();
  for (int kt = 0; kt < nk; ++kt) {
    const int buf = kt & 1;
    const bool more = (kt + 1) < nk;
    if (more) loadTiles(kt + 1);       // global loads overlap with WMMA below
    if (kt + 2 < nk) prefetchTiles(kt + 2);
    compute(buf);
    if (more) {
      __syncthreads();
      storeTiles(buf ^ 1);
      __syncthreads();
    }
  }

  // Epilogue. C/D layout: VGPR r, lanes 0-15 -> (M=r, N=lane); lanes 16-31 ->
  // (M=r+8, N=lane-16).
#pragma unroll
  for (int a = 0; a < 4; ++a)
#pragma unroll
    for (int b = 0; b < 4; ++b)
#pragma unroll
      for (int r = 0; r < 8; ++r) {
        const int row = m0 + wm * 64 + a * 16 + r + (hi ? 8 : 0);
        const int col = n0 + wn * 64 + b * 16 + l15;
        float v = acc[a][b][r];
        if constexpr (EPI == 2 || EPI == 3) v += bias[col];
        if constexpr (EPI == 2)
          v = 0.5f * v * (1.0f + erff(v * 0.70710678118654752f)); // exact GELU
        if constexpr (EPI == 3) v += resid[(size_t)row * N + col];
        C[(size_t)row * N + col] = v;
      }
}

// ---------------------------------------------------------------------------
// Flash attention (causal), bf16 WMMA, online softmax.
// Grid: (SEQ/128, NH, BS). Block: 256 threads = 8 waves; wave w owns 16 query
// rows. Key blocks of 32 (one WMMA K step for P*V).
// qkv layout: [b, s, 3072], head h at column h*384 (+0 q, +128 k, +256 v).
// ---------------------------------------------------------------------------
__global__ __launch_bounds__(256)
void attn_wmma_bf16(const float* __restrict__ qkv, float* __restrict__ ctx)
{
  __shared__ unsigned short qs[128][136];  // [qrow][hd]      34816 B
  __shared__ unsigned short ks[32][136];   // [krow][hd]       8704 B
  __shared__ unsigned short vs[128][40];   // [hd][krow] (V^T) 10240 B
  __shared__ unsigned short ps[128][40];   // [qrow][krel]     10240 B

  const int tid  = threadIdx.x;
  const int wave = tid >> 5;
  const int lane = tid & 31;
  const int l15  = lane & 15;
  const bool hi  = lane >= 16;
  const int qb   = blockIdx.x * 128;
  const int h    = blockIdx.y;
  const int b    = blockIdx.z;

  const float* base = qkv + (size_t)b * SEQ * QKVD + h * (3 * HD);

  // Load Q tile (128 x 128 f32 -> bf16 LDS)
#pragma unroll
  for (int i = 0; i < 16; ++i) {
    const int f = tid + 256 * i;                // 0..4095
    const int row = f >> 5, c0 = (f & 31) * 4;
    float4 t = *(const float4*)(base + (size_t)(qb + row) * QKVD + c0);
    uint2 p; p.x = pk2(t.x, t.y); p.y = pk2(t.z, t.w);
    *(uint2*)&qs[row][c0] = p;
  }

  float m_[8], l_[8];
  v8f cacc[8];
#pragma unroll
  for (int r = 0; r < 8; ++r) { m_[r] = -1e30f; l_[r] = 0.0f; }
#pragma unroll
  for (int t = 0; t < 8; ++t)
#pragma unroll
    for (int r = 0; r < 8; ++r) cacc[t][r] = 0.0f;

  const float scale = 0.08838834764831845f;    // 1/sqrt(128)
  const int njb = qb / 32 + 4;                 // causal: only blocks <= diagonal

  for (int jb = 0; jb < njb; ++jb) {
    const int kb = jb * 32;
    __syncthreads();                           // prev iter done with ks/vs/ps
    // Load K block (32 x 128) and V block transposed (-> [hd][krow])
#pragma unroll
    for (int i = 0; i < 4; ++i) {
      const int f = tid + 256 * i;             // 0..1023
      const int row = f >> 5, c0 = (f & 31) * 4;
      float4 tk = *(const float4*)(base + HD + (size_t)(kb + row) * QKVD + c0);
      uint2 p; p.x = pk2(tk.x, tk.y); p.y = pk2(tk.z, tk.w);
      *(uint2*)&ks[row][c0] = p;
      float4 tv = *(const float4*)(base + 2 * HD + (size_t)(kb + row) * QKVD + c0);
      vs[c0 + 0][row] = f2bf(tv.x);
      vs[c0 + 1][row] = f2bf(tv.y);
      vs[c0 + 2][row] = f2bf(tv.z);
      vs[c0 + 3][row] = f2bf(tv.w);
    }
    __syncthreads();

    // S = Q * K^T  (16 qrows x 32 keys per wave), K dim = HD in 4 steps of 32
    v8f sacc[2];
#pragma unroll
    for (int t = 0; t < 2; ++t)
#pragma unroll
      for (int r = 0; r < 8; ++r) sacc[t][r] = 0.0f;
#pragma unroll
    for (int kk = 0; kk < 4; ++kk) {
      F16x fa;
      const int row = 16 * wave + l15;
      const int k0  = 32 * kk + (hi ? 8 : 0);
      fa.q[0] = *(const uint4*)&qs[row][k0];
      fa.q[1] = *(const uint4*)&qs[row][k0 + 16];
#pragma unroll
      for (int t = 0; t < 2; ++t) {
        F16x fb;
        const int col = 16 * t + l15;          // key row
        const int kb0 = 32 * kk + (hi ? 8 : 0);
        fb.q[0] = *(const uint4*)&ks[col][kb0];
        fb.q[1] = *(const uint4*)&ks[col][kb0 + 16];
        sacc[t] = wmma_bf16(fa, fb, sacc[t]);
      }
    }

    // Online softmax over the 16x32 slab (row stats reduce across the 16
    // lanes of each half-wave, matching the C layout).
#pragma unroll
    for (int r = 0; r < 8; ++r) {
      const int grow = qb + 16 * wave + r + (hi ? 8 : 0);
      float v0 = sacc[0][r] * scale;
      float v1 = sacc[1][r] * scale;
      if (kb + l15 > grow)      v0 = -1e30f;   // causal mask
      if (kb + 16 + l15 > grow) v1 = -1e30f;
      float mx = fmaxf(v0, v1);
#pragma unroll
      for (int o = 1; o < 16; o <<= 1) mx = fmaxf(mx, __shfl_xor(mx, o, 32));
      const float mnew = fmaxf(m_[r], mx);
      const float so   = __expf(m_[r] - mnew);
      const float p0   = __expf(v0 - mnew);
      const float p1   = __expf(v1 - mnew);
      float rs = p0 + p1;
#pragma unroll
      for (int o = 1; o < 16; o <<= 1) rs += __shfl_xor(rs, o, 32);
      l_[r] = l_[r] * so + rs;
      m_[r] = mnew;
#pragma unroll
      for (int t = 0; t < 8; ++t) cacc[t][r] *= so;   // rescale running ctx
      const int prow = 16 * wave + r + (hi ? 8 : 0);
      ps[prow][l15]      = f2bf(p0);
      ps[prow][16 + l15] = f2bf(p1);
    }
    __syncthreads();

    // ctx += P * V  (A = P from LDS, B = V^T-layout fragments)
    F16x fp;
    const int prow = 16 * wave + l15;
    const int pk0  = hi ? 8 : 0;
    fp.q[0] = *(const uint4*)&ps[prow][pk0];
    fp.q[1] = *(const uint4*)&ps[prow][pk0 + 16];
#pragma unroll
    for (int t = 0; t < 8; ++t) {
      F16x fv;
      const int vc  = 16 * t + l15;            // head-dim column
      const int vk0 = hi ? 8 : 0;
      fv.q[0] = *(const uint4*)&vs[vc][vk0];
      fv.q[1] = *(const uint4*)&vs[vc][vk0 + 16];
      cacc[t] = wmma_bf16(fp, fv, cacc[t]);
    }
  }

  // Normalize and write ctx[b, row, h*128 + d]
#pragma unroll
  for (int r = 0; r < 8; ++r) {
    const float inv = 1.0f / l_[r];
    const int row = qb + 16 * wave + r + (hi ? 8 : 0);
#pragma unroll
    for (int t = 0; t < 8; ++t) {
      const int col = h * HD + 16 * t + l15;
      ctx[((size_t)b * SEQ + row) * DM + col] = cacc[t][r] * inv;
    }
  }
}

// ---------------------------------------------------------------------------
// LayerNorm over D=1024, one block (256 threads) per token row.
// ---------------------------------------------------------------------------
__global__ __launch_bounds__(256)
void layernorm_kernel(const float* __restrict__ x, const float* __restrict__ g,
                      const float* __restrict__ b, float* __restrict__ y)
{
  const int row = blockIdx.x;
  const int tid = threadIdx.x;
  const float* xr = x + (size_t)row * DM;
  float v[4]; float s = 0.0f, s2 = 0.0f;
#pragma unroll
  for (int i = 0; i < 4; ++i) {
    v[i] = xr[tid + 256 * i];
    s  += v[i];
    s2 += v[i] * v[i];
  }
#pragma unroll
  for (int o = 16; o >= 1; o >>= 1) {
    s  += __shfl_xor(s,  o, 32);
    s2 += __shfl_xor(s2, o, 32);
  }
  __shared__ float ss[8], ss2[8];
  if ((tid & 31) == 0) { ss[tid >> 5] = s; ss2[tid >> 5] = s2; }
  __syncthreads();
  float ts = 0.0f, ts2 = 0.0f;
#pragma unroll
  for (int w = 0; w < 8; ++w) { ts += ss[w]; ts2 += ss2[w]; }
  const float mean = ts * (1.0f / DM);
  const float var  = ts2 * (1.0f / DM) - mean * mean;
  const float rstd = rsqrtf(var + 1e-5f);
  float* yr = y + (size_t)row * DM;
#pragma unroll
  for (int i = 0; i < 4; ++i) {
    const int c = tid + 256 * i;
    yr[c] = (v[i] - mean) * rstd * g[c] + b[c];
  }
}

// ---------------------------------------------------------------------------
// Host-side orchestration (graph-capture safe: only kernel launches on stream)
// ---------------------------------------------------------------------------
extern "C" void kernel_launch(void* const* d_in, const int* in_sizes, int n_in,
                              void* d_out, int out_size, void* d_ws, size_t ws_size,
                              hipStream_t stream)
{
  (void)in_sizes; (void)n_in; (void)out_size; (void)ws_size;

  const float* x     = (const float*)d_in[0];
  const float* Wqkv  = (const float*)d_in[1];
  const float* Wout  = (const float*)d_in[2];
  const float* b_out = (const float*)d_in[3];
  const float* ln_g  = (const float*)d_in[4];
  const float* ln_b  = (const float*)d_in[5];
  const float* W1    = (const float*)d_in[6];
  const float* b1    = (const float*)d_in[7];
  const float* W2    = (const float*)d_in[8];
  const float* b2    = (const float*)d_in[9];
  float* out = (float*)d_out;                       // y and final result

  // Workspace layout (floats): qkv | ctx | h | ffn1
  float* qkv  = (float*)d_ws;                       // 4096 x 3072
  float* ctxb = qkv  + (size_t)MTOT * QKVD;         // 4096 x 1024
  float* hbuf = ctxb + (size_t)MTOT * DM;           // 4096 x 1024
  float* gbuf = hbuf + (size_t)MTOT * DM;           // 4096 x 4096

  const dim3 blk(256);

  // 1) qkv = x @ Wqkv
  gemm_wmma_bf16<0><<<dim3(QKVD / 256, MTOT / 128), blk, 0, stream>>>(
      x, Wqkv, nullptr, nullptr, qkv, MTOT, QKVD, DM);

  // 2) ctx = causal_softmax(q k^T / sqrt(hd)) v
  attn_wmma_bf16<<<dim3(SEQ / 128, NH, BS), blk, 0, stream>>>(qkv, ctxb);

  // 3) y = x + ctx @ Wout + b_out   (y kept in d_out)
  gemm_wmma_bf16<3><<<dim3(DM / 256, MTOT / 128), blk, 0, stream>>>(
      ctxb, Wout, b_out, x, out, MTOT, DM, DM);

  // 4) h = LN(y)
  layernorm_kernel<<<dim3(MTOT), blk, 0, stream>>>(out, ln_g, ln_b, hbuf);

  // 5) g = GELU(h @ W1 + b1)
  gemm_wmma_bf16<2><<<dim3(FFN / 256, MTOT / 128), blk, 0, stream>>>(
      hbuf, W1, b1, nullptr, gbuf, MTOT, FFN, DM);

  // 6) out = y + g @ W2 + b2  (in place on d_out; each element read then written
  //    exactly once by the same thread)
  gemm_wmma_bf16<3><<<dim3(DM / 256, MTOT / 128), blk, 0, stream>>>(
      gbuf, W2, b2, out, out, MTOT, DM, FFN);
}
